// Net_3350074490963
// MI455X (gfx1250) — matrix operand order repaired and checked by reading, not compiled, not run on hardware
//
#include <hip/hip_runtime.h>
#include <stdint.h>

typedef __attribute__((ext_vector_type(16))) _Float16 v16h;
typedef __attribute__((ext_vector_type(8)))  float    v8f;

#define NN   100000
#define NL   4
#define NGR  512
#define NOUT 10

union Frag { unsigned u[8]; v16h h; };

__device__ __forceinline__ unsigned pack2(float a, float b) {
  unsigned short ua = __builtin_bit_cast(unsigned short, (_Float16)a);
  unsigned short ub = __builtin_bit_cast(unsigned short, (_Float16)b);
  return (unsigned)ua | ((unsigned)ub << 16);
}

// ---- prep: pack f32 weights into WMMA B-fragment order (f16, per layer/tile/kstep) ----
// linear index = (((l*NT + t)*NS + s)*32 + lane)*8 + v ; halves: K = s*32 + 16g + 2v + h, N = t*16 + lane%16
__global__ void pack_w_kernel(const float* __restrict__ w, unsigned* __restrict__ out,
                              int K, int N, int NT, int NS, int total) {
  int tid = blockIdx.x * blockDim.x + threadIdx.x;
  if (tid >= total) return;
  int v = tid & 7, lane = (tid >> 3) & 31;
  int q = tid >> 8;
  int s = q % NS; q /= NS;
  int t = q % NT; int l = q / NT;
  int g = lane >> 4;
  int k = s * 32 + g * 16 + 2 * v;
  int n = t * 16 + (lane & 15);
  const float* wl = w + (size_t)l * K * N;
  out[tid] = pack2(wl[(size_t)k * N + n], wl[(size_t)(k + 1) * N + n]);
}

// fold bias+BN into y = z*A + C
__global__ void bn_prep_kernel(const float* g, const float* b, const float* m,
                               const float* v, const float* lb, float* A, float* C, int n) {
  int i = blockIdx.x * blockDim.x + threadIdx.x;
  if (i >= n) return;
  float a = g[i] * rsqrtf(v[i] + 1e-5f);
  A[i] = a;
  C[i] = (lb[i] - m[i]) * a + b[i];
}

// ---- edge aggregation: agg[dst] += x[src]; one wave32 per edge, float4 per lane ----
__global__ void agg_kernel(const float* __restrict__ x, const int* __restrict__ src,
                           const int* __restrict__ dst, float* __restrict__ agg, int ne) {
  int e = blockIdx.x * 8 + (threadIdx.x >> 5);
  if (e >= ne) return;
  int lane = threadIdx.x & 31;
  int s = src[e], d = dst[e];
  // get the next edge-group's gather row moving toward this WGP early
  int en = (e + 8 < ne) ? (e + 8) : e;
  __builtin_prefetch(x + (size_t)src[en] * 128 + lane * 4, 0, 1);
  float4 vv = *reinterpret_cast<const float4*>(x + (size_t)s * 128 + lane * 4);
  float* ap = agg + (size_t)d * 128 + lane * 4;
  unsafeAtomicAdd(ap + 0, vv.x);
  unsafeAtomicAdd(ap + 1, vv.y);
  unsafeAtomicAdd(ap + 2, vv.z);
  unsafeAtomicAdd(ap + 3, vv.w);
}

// ---- fused GIN layer: (1+eps)*x + agg -> GEMM1(128->256) BN ReLU -> GEMM2(256->128) BN ReLU ----
// block = 256 threads (8 waves), 32 node rows (two M-tiles) per block; every B fragment is
// reused across both M-tiles, halving L2 weight traffic. WMMA f16 in, f32 accumulate.
__global__ void __launch_bounds__(256)
gin_layer_kernel(const float* __restrict__ x_in, const float* __restrict__ agg,
                 const float* __restrict__ eps_all, int layer,
                 const unsigned* __restrict__ w1p, const unsigned* __restrict__ w2p,
                 const float* __restrict__ A1, const float* __restrict__ C1,
                 const float* __restrict__ A2, const float* __restrict__ C2,
                 float* __restrict__ x_out) {
  __shared__ __align__(16) _Float16 sh_in[32 * 136];   // 32 x 128 (+8 pad halves)
  __shared__ __align__(16) _Float16 sh_mid[32 * 264];  // 32 x 256 (+8 pad halves)

  const int tid = threadIdx.x;
  const int row0 = blockIdx.x * 32;   // NN % 32 == 0
  const float epl = 1.0f + eps_all[layer];

  // stage fused input tile, f32 -> f16
  for (int idx = tid; idx < 32 * 128; idx += 256) {
    int m = idx >> 7, k = idx & 127;
    size_t gi = (size_t)(row0 + m) * 128 + k;
    sh_in[m * 136 + k] = (_Float16)(epl * x_in[gi] + agg[gi]);
  }
  __syncthreads();

  const int wave = tid >> 5, lane = tid & 31;
  const int g = lane >> 4, mr = lane & 15;

  // A fragments for GEMM1 (K = 128 -> 4 k-steps) for both M-tiles, contiguous k = 16g + i order
  const unsigned* shin32 = reinterpret_cast<const unsigned*>(sh_in);
  Frag af[2][4];
#pragma unroll
  for (int mt = 0; mt < 2; ++mt)
#pragma unroll
    for (int s = 0; s < 4; ++s)
#pragma unroll
      for (int v = 0; v < 8; ++v) {
        int K0 = 2 * v + 16 * g;
        af[mt][s].u[v] = shin32[(mt * 16 + mr) * 68 + s * 16 + (K0 >> 1)];
      }

  const float* A1l = A1 + layer * 256;
  const float* C1l = C1 + layer * 256;
#pragma unroll
  for (int t = 0; t < 2; ++t) {        // 16 N-tiles over 8 waves -> 2 per wave
    int tindex = wave * 2 + t;
    v8f acc0 = {}, acc1 = {};
#pragma unroll
    for (int s = 0; s < 4; ++s) {
      Frag bf;
      bf.h = *reinterpret_cast<const v16h*>(
          w1p + (((((size_t)layer * 16 + tindex) * 4 + s) * 32 + lane) * 8));
      acc0 = __builtin_amdgcn_wmma_f32_16x16x32_f16(false, af[0][s].h, false, bf.h,
                                                    (short)0, acc0, false, false);
      acc1 = __builtin_amdgcn_wmma_f32_16x16x32_f16(false, af[1][s].h, false, bf.h,
                                                    (short)0, acc1, false, false);
    }
    int col = tindex * 16 + mr;
    float sc = A1l[col], sh = C1l[col];
#pragma unroll
    for (int i = 0; i < 8; ++i) {      // C elem i -> row i+8g, col lane%16
      float y0 = fmaxf(fmaf(acc0[i], sc, sh), 0.0f);
      float y1 = fmaxf(fmaf(acc1[i], sc, sh), 0.0f);
      sh_mid[(i + 8 * g) * 264 + col]      = (_Float16)y0;
      sh_mid[(16 + i + 8 * g) * 264 + col] = (_Float16)y1;
    }
  }
  __syncthreads();

  // GEMM2: K = 256 -> 8 k-steps, one 16-col N-tile per wave, both M-tiles
  const unsigned* shmid32 = reinterpret_cast<const unsigned*>(sh_mid);
  v8f acc0 = {}, acc1 = {};
#pragma unroll
  for (int s = 0; s < 8; ++s) {
    Frag a0, a1, bf;
#pragma unroll
    for (int v = 0; v < 8; ++v) {
      int K0 = 2 * v + 16 * g;
      a0.u[v] = shmid32[mr * 132 + s * 16 + (K0 >> 1)];
      a1.u[v] = shmid32[(16 + mr) * 132 + s * 16 + (K0 >> 1)];
    }
    bf.h = *reinterpret_cast<const v16h*>(
        w2p + (((((size_t)layer * 8 + wave) * 8 + s) * 32 + lane) * 8));
    acc0 = __builtin_amdgcn_wmma_f32_16x16x32_f16(false, a0.h, false, bf.h,
                                                  (short)0, acc0, false, false);
    acc1 = __builtin_amdgcn_wmma_f32_16x16x32_f16(false, a1.h, false, bf.h,
                                                  (short)0, acc1, false, false);
  }
  int col2 = wave * 16 + mr;
  float sc2 = A2[layer * 128 + col2], sh2 = C2[layer * 128 + col2];
#pragma unroll
  for (int i = 0; i < 8; ++i) {
    float y0 = fmaxf(fmaf(acc0[i], sc2, sh2), 0.0f);
    float y1 = fmaxf(fmaf(acc1[i], sc2, sh2), 0.0f);
    x_out[(size_t)(row0 + i + 8 * g) * 128 + col2]      = y0;
    x_out[(size_t)(row0 + 16 + i + 8 * g) * 128 + col2] = y1;
  }
}

// ---- graph pooling: pooled[batch[n]] += x[n] ----
__global__ void pool_kernel(const float* __restrict__ x, const int* __restrict__ batch,
                            float* __restrict__ pooled, int total) {
  int t = blockIdx.x * blockDim.x + threadIdx.x;
  if (t >= total) return;
  int node = t >> 7, c = t & 127;
  unsafeAtomicAdd(pooled + (size_t)batch[node] * 128 + c, x[t]);
}

// ---- head: lin1 + BN3 + ReLU + lin2 + log_softmax ; one block (128 thr) per graph ----
__global__ void head_kernel(const float* __restrict__ pooled, const float* __restrict__ lin1_w,
                            const float* __restrict__ A3, const float* __restrict__ C3,
                            const float* __restrict__ lin2_w, const float* __restrict__ lin2_b,
                            float* __restrict__ out) {
  __shared__ float row[128];
  __shared__ float hbuf[128];
  __shared__ float lg[16];
  int gph = blockIdx.x, j = threadIdx.x;
  row[j] = pooled[(size_t)gph * 128 + j];
  __syncthreads();
  float acc = 0.f;
#pragma unroll 4
  for (int k = 0; k < 128; ++k) acc = fmaf(row[k], lin1_w[k * 128 + j], acc);
  hbuf[j] = fmaxf(fmaf(acc, A3[j], C3[j]), 0.f);
  __syncthreads();
  if (j < NOUT) {
    float a = lin2_b[j];
    for (int k = 0; k < 128; ++k) a = fmaf(hbuf[k], lin2_w[k * NOUT + j], a);
    lg[j] = a;
  }
  __syncthreads();
  if (j < NOUT) {
    float mx = -1e30f;
    for (int o = 0; o < NOUT; ++o) mx = fmaxf(mx, lg[o]);
    float ssum = 0.f;
    for (int o = 0; o < NOUT; ++o) ssum += expf(lg[o] - mx);
    out[gph * NOUT + j] = lg[j] - mx - logf(ssum);
  }
}

extern "C" void kernel_launch(void* const* d_in, const int* in_sizes, int n_in,
                              void* d_out, int out_size, void* d_ws, size_t ws_size,
                              hipStream_t stream) {
  const float* x     = (const float*)d_in[0];
  const int*   ei    = (const int*)d_in[1];
  const int*   batch = (const int*)d_in[2];
  const float* w1    = (const float*)d_in[3];
  const float* b1    = (const float*)d_in[4];
  const float* bn1g  = (const float*)d_in[5];
  const float* bn1b  = (const float*)d_in[6];
  const float* bn1m  = (const float*)d_in[7];
  const float* bn1v  = (const float*)d_in[8];
  const float* eps   = (const float*)d_in[9];
  const float* w2    = (const float*)d_in[10];
  const float* b2    = (const float*)d_in[11];
  const float* bn2g  = (const float*)d_in[12];
  const float* bn2b  = (const float*)d_in[13];
  const float* bn2m  = (const float*)d_in[14];
  const float* bn2v  = (const float*)d_in[15];
  const float* l1w   = (const float*)d_in[16];
  const float* l1b   = (const float*)d_in[17];
  const float* bn3g  = (const float*)d_in[18];
  const float* bn3b  = (const float*)d_in[19];
  const float* bn3m  = (const float*)d_in[20];
  const float* bn3v  = (const float*)d_in[21];
  const float* l2w   = (const float*)d_in[22];
  const float* l2b   = (const float*)d_in[23];
  float* out = (float*)d_out;

  const int n_edges = in_sizes[1] / 2;
  (void)n_in; (void)out_size; (void)ws_size;

  char* ws = (char*)d_ws;
  size_t off = 0;
  auto wsAlloc = [&](size_t bytes) -> void* {
    void* p = ws + off;
    off = (off + bytes + 255) & ~(size_t)255;
    return p;
  };
  float*    xb0    = (float*)wsAlloc((size_t)NN * 128 * 4);
  float*    xb1    = (float*)wsAlloc((size_t)NN * 128 * 4);
  float*    agg    = (float*)wsAlloc((size_t)NN * 128 * 4);
  float*    pooled = (float*)wsAlloc((size_t)NGR * 128 * 4);
  unsigned* w1p    = (unsigned*)wsAlloc(65536 * 4);
  unsigned* w2p    = (unsigned*)wsAlloc(65536 * 4);
  float*    A1     = (float*)wsAlloc(NL * 256 * 4);
  float*    C1     = (float*)wsAlloc(NL * 256 * 4);
  float*    A2     = (float*)wsAlloc(NL * 128 * 4);
  float*    C2     = (float*)wsAlloc(NL * 128 * 4);
  float*    A3     = (float*)wsAlloc(128 * 4);
  float*    C3     = (float*)wsAlloc(128 * 4);

  // one-time (per launch) prep: weight packing + BN/bias folding
  pack_w_kernel<<<65536 / 256, 256, 0, stream>>>(w1, w1p, 128, 256, 16, 4, 65536);
  pack_w_kernel<<<65536 / 256, 256, 0, stream>>>(w2, w2p, 256, 128, 8, 8, 65536);
  bn_prep_kernel<<<4, 256, 0, stream>>>(bn1g, bn1b, bn1m, bn1v, b1, A1, C1, NL * 256);
  bn_prep_kernel<<<2, 256, 0, stream>>>(bn2g, bn2b, bn2m, bn2v, b2, A2, C2, NL * 128);
  bn_prep_kernel<<<1, 128, 0, stream>>>(bn3g, bn3b, bn3m, bn3v, l1b, A3, C3, 128);

  float* bufs[2] = { xb0, xb1 };
  for (int l = 0; l < NL; ++l) {
    const float* cur = (l == 0) ? x : bufs[(l + 1) & 1];
    float* nxt = bufs[l & 1];
    hipMemsetAsync(agg, 0, (size_t)NN * 128 * 4, stream);
    agg_kernel<<<(n_edges + 7) / 8, 256, 0, stream>>>(cur, ei, ei + n_edges, agg, n_edges);
    gin_layer_kernel<<<NN / 32, 256, 0, stream>>>(cur, agg, eps, l, w1p, w2p,
                                                  A1, C1, A2, C2, nxt);
  }

  hipMemsetAsync(pooled, 0, (size_t)NGR * 128 * 4, stream);
  pool_kernel<<<(NN * 128 + 255) / 256, 256, 0, stream>>>(bufs[(NL + 1) & 1], batch,
                                                          pooled, NN * 128);
  head_kernel<<<NGR, 128, 0, stream>>>(pooled, l1w, A3, C3, l2w, l2b, out);
}
